// HWconv2d_14611478741629
// MI455X (gfx1250) — matrix operand order repaired and checked by reading, not compiled, not run on hardware
//
#include <hip/hip_runtime.h>

typedef _Float16 v16h __attribute__((ext_vector_type(16)));
typedef _Float16 v8h  __attribute__((ext_vector_type(8)));
typedef float    v8f  __attribute__((ext_vector_type(8)));

#define CIN   64
#define COUT  128
#define HH    56
#define WW    56
#define BB    32
#define KSTEPS 18            // 576 / 32
#define MTILES 8             // 128 / 16
#define XW_N   58            // W + 2 halo
#define XW_STRIDE 72         // padded ci-slot stride in halfs (36 dwords -> bank-conflict-free)
#define LDS_HALF (3*XW_N*XW_STRIDE)   // 12528 halfs per (hi|lo) plane

// ---------------------------------------------------------------------------
// e5m2 (BF8) quantization, qtorch semantics: RNE on 2-bit mantissa grid,
// subnormals flushed to the 2^-16 grid, clamp to +-57344, signed zero -> 0.
// Result is exactly representable in f16.
// ---------------------------------------------------------------------------
__device__ __forceinline__ float quant_e5m2(float v) {
    float av = __builtin_fabsf(v);
    if (!(av > 0.0f)) return 0.0f;
    int e = (int)((__float_as_uint(av) >> 23) & 0xFF) - 127;  // floor(log2|v|)
    if (e < -14) e = -14;                                     // denormal grid
    int e2 = e - 2;                                           // man = 2
    float scale  = __uint_as_float((unsigned)(e2 + 127) << 23);
    float iscale = __uint_as_float((unsigned)(127 - e2) << 23);
    float q = rintf(v * iscale) * scale;                      // RNE
    q = fminf(fmaxf(q, -57344.0f), 57344.0f);
    return q;
}

// ---------------------------------------------------------------------------
// Kernel 1: quantize weights and pack into A-matrix fragment order.
// k index = (kh*3+kw)*64 + ci.  A-frag element e for lane (m=lane%16,h=lane/16):
//   k = kk*32 + 16h + e   (contiguous k order, matches B fragment)
// wp[((mtile*18+kk)*32+lane)*16 + e] = f16(quant(w[m][ci][kh][kw]))
// ---------------------------------------------------------------------------
__global__ void pack_weights(const float* __restrict__ w, _Float16* __restrict__ wp) {
    int idx = blockIdx.x * blockDim.x + threadIdx.x;
    if (idx >= MTILES * KSTEPS * 32 * 16) return;
    int e    = idx & 15;
    int lane = (idx >> 4) & 31;
    int kk   = (idx >> 9) % KSTEPS;
    int mt   = idx / (KSTEPS * 512);
    int h    = lane >> 4;
    int m    = mt * 16 + (lane & 15);
    int k    = kk * 32 + 16 * h + e;
    int f    = k >> 6;
    int ci   = k & 63;
    int kh   = f / 3, kw = f - 3 * kh;
    float wv = w[((m * CIN + ci) * 3 + kh) * 3 + kw];
    wp[idx] = (_Float16)quant_e5m2(wv);
}

__device__ __forceinline__ v16h ld_frag_lds(const _Float16* p) {
    union { v16h v; v8h h[2]; } u;
    u.h[0] = *(const v8h*)(p);       // ds_load_b128
    u.h[1] = *(const v8h*)(p + 8);   // ds_load_b128
    return u.v;
}

// ---------------------------------------------------------------------------
// Kernel 2: one block per (batch, output row). 256 threads = 8 waves.
// Wave m computes Cout rows 16m..16m+15 for all 64 (padded) pixels of the row.
// x is staged in LDS as hi/lo f16 planes [ry][xw][ci] (ci contiguous).
// Two WMMAs per k-step (hi + lo) reconstruct fp32-input numerics.
// ---------------------------------------------------------------------------
__global__ __launch_bounds__(256) void conv_wmma(
    const float* __restrict__ x, const _Float16* __restrict__ wp,
    const float* __restrict__ bias, float* __restrict__ out) {

    __shared__ _Float16 sh[2 * LDS_HALF];
    _Float16* xs_hi = sh;
    _Float16* xs_lo = sh + LDS_HALF;

    const int bid = blockIdx.x;
    const int b   = bid / HH;
    const int h0  = bid % HH;
    const int tid = threadIdx.x;

    // ---- stage 3 halo rows of x into LDS as f16 hi/lo (coalesced in gx) ----
    const int NF = 3 * XW_N * CIN;           // 11136 elements
    for (int idx = tid; idx < NF; idx += 256) {
        int xw = idx % XW_N;
        int r  = idx / XW_N;
        int ry = r % 3;
        int ci = r / 3;
        int gy = h0 - 1 + ry;
        int gx = xw - 1;
        float v = 0.0f;
        if (gy >= 0 && gy < HH && gx >= 0 && gx < WW)
            v = x[((b * CIN + ci) * HH + gy) * WW + gx];
        _Float16 hv = (_Float16)v;
        _Float16 lv = (_Float16)(v - (float)hv);
        int off = (ry * XW_N + xw) * XW_STRIDE + ci;
        xs_hi[off] = hv;
        xs_lo[off] = lv;
    }
    __syncthreads();

    const int lane = tid & 31;
    const int wid  = tid >> 5;     // wave id == M tile
    const int hsel = lane >> 4;
    const int nlo  = lane & 15;

    v8f zero;
    #pragma unroll
    for (int i = 0; i < 8; ++i) zero[i] = 0.0f;
    v8f acc[4] = { zero, zero, zero, zero };

    int np[4];                      // clamped pixel per N-tile (keep EXEC full)
    #pragma unroll
    for (int nt = 0; nt < 4; ++nt) {
        int n = nt * 16 + nlo;
        np[nt] = (n > WW - 1) ? (WW - 1) : n;
    }

    const _Float16* wbase = wp + (size_t)(wid * KSTEPS) * 512;

    for (int kk = 0; kk < KSTEPS; ++kk) {
        // A fragment: prepacked, 32B/lane contiguous -> 2x global_load_b128
        v16h a = *(const v16h*)(wbase + ((size_t)kk * 32 + lane) * 16);

        // B fragment source: lane needs k = s..s+15, s = kk*32 + 16*hsel
        int s    = kk * 32 + (hsel << 4);
        int f    = s >> 6;          // filter tap (kh*3+kw), 0..8
        int ci0  = s & 63;          // 16-aligned, never crosses tap boundary
        int ry   = f / 3;
        int kw   = f - 3 * ry;
        int rowb = (ry * XW_N) * XW_STRIDE + ci0;

        #pragma unroll
        for (int nt = 0; nt < 4; ++nt) {
            int off = rowb + (np[nt] + kw) * XW_STRIDE;
            v16h bh = ld_frag_lds(xs_hi + off);
            v16h bl = ld_frag_lds(xs_lo + off);
            acc[nt] = __builtin_amdgcn_wmma_f32_16x16x32_f16(
                false, a, false, bh, (short)0, acc[nt], false, false);
            acc[nt] = __builtin_amdgcn_wmma_f32_16x16x32_f16(
                false, a, false, bl, (short)0, acc[nt], false, false);
        }
    }

    // ---- epilogue: +72*bias, ReLU, store (D layout: VGPR r -> M = r + 8*hsel)
    float bs[8];
    #pragma unroll
    for (int r = 0; r < 8; ++r)
        bs[r] = 72.0f * bias[wid * 16 + hsel * 8 + r];

    #pragma unroll
    for (int nt = 0; nt < 4; ++nt) {
        int n = nt * 16 + nlo;
        if (n < WW) {
            size_t base = (((size_t)b * COUT + wid * 16 + hsel * 8) * HH + h0) * WW + n;
            #pragma unroll
            for (int r = 0; r < 8; ++r) {
                float o = acc[nt][r] + bs[r];
                out[base + (size_t)r * (HH * WW)] = fmaxf(o, 0.0f);
            }
        }
    }
}

// ---------------------------------------------------------------------------
extern "C" void kernel_launch(void* const* d_in, const int* in_sizes, int n_in,
                              void* d_out, int out_size, void* d_ws, size_t ws_size,
                              hipStream_t stream) {
    (void)in_sizes; (void)n_in; (void)out_size; (void)ws_size;
    const float* x    = (const float*)d_in[0];
    const float* w    = (const float*)d_in[1];
    const float* bias = (const float*)d_in[2];
    float* out        = (float*)d_out;
    _Float16* wp      = (_Float16*)d_ws;   // 73728 halfs = 144 KB packed weights

    pack_weights<<<(MTILES * KSTEPS * 32 * 16) / 256, 256, 0, stream>>>(w, wp);
    conv_wmma<<<BB * HH, 256, 0, stream>>>(x, wp, bias, out);
}
